// TreeRCNN_11484742550034
// MI455X (gfx1250) — compile-verified, matrix operand order checked by reading, not run on hardware
//
#include <hip/hip_runtime.h>

typedef __attribute__((ext_vector_type(16))) _Float16 v16h;
typedef __attribute__((ext_vector_type(8)))  float    v8f;

#define MAX_PTS 512
#define MIN_PTS 4
#define NWAVES 8
#define BLOCK 256
#define F1_STRIDE 72    // 64 + 8 halves pad = 144 B rows (16B aligned, bank-spread)
#define W2T_STRIDE 72

struct __align__(16) Smem {
  union {
    int selW[NWAVES][MAX_PTS];        // phase 1: per-wave ordered index lists (16384 B)
    _Float16 f1[256 * F1_STRIDE];     // phase 2: F1 tile for one M-half (36864 B)
  } u;
  _Float16 w2t[128 * W2T_STRIDE];     // W2^T in f16 (18432 B)
  float px[MAX_PTS], py[MAX_PTS], pz[MAX_PTS]; // centered gathered points (6144 B)
  unsigned wcnt[NWAVES];
  unsigned wbase[NWAVES];
  unsigned gmax[128];                 // column max as float bits (relu >= 0)
  float hpart[32 * 5];
  unsigned total;
};                                     // ~62.7 KB total

__device__ __forceinline__ v16h ld_frag(const _Float16* p) {
  // v16h elements 0..7 <- p[0..7], 8..15 <- p[16..23]  (ISA 16-bit A/B layout)
  union { v16h h; float4 f[2]; } t;
  t.f[0] = *(const float4*)(p);
  t.f[1] = *(const float4*)(p + 16);
  return t.h;
}

__global__ __launch_bounds__(BLOCK) void treercnn_kernel(
    const float* __restrict__ points, const float* __restrict__ anchors,
    const float* __restrict__ W1, const float* __restrict__ b1,
    const float* __restrict__ W2, const float* __restrict__ b2,
    const float* __restrict__ Wc, const float* __restrict__ bc,
    const float* __restrict__ Wr, const float* __restrict__ br,
    float* __restrict__ out, int N) {
  __shared__ Smem s;
  const int tid  = threadIdx.x;
  const int lane = tid & 31;
  const int wave = tid >> 5;
  const int anc  = blockIdx.x;

  // anchor box bounds
  const float ax = anchors[anc*6+0], ay = anchors[anc*6+1];
  const float aw = anchors[anc*6+3], al = anchors[anc*6+4], ah = anchors[anc*6+5];
  const float x0 = ax - aw*0.5f, x1 = ax + aw*0.5f;
  const float y0 = ay - al*0.5f, y1 = ay + al*0.5f;

  if (tid < 128) s.gmax[tid] = 0u;

  // W2^T -> f16 LDS: w2t[n][k] = W2[k][n]  (coalesced global reads over n)
  for (int idx = tid; idx < 64*128; idx += BLOCK) {
    int n = idx & 127, k = idx >> 7;
    s.w2t[n*W2T_STRIDE + k] = (_Float16)W2[k*128 + n];
  }

  // ---- Phase 1: per-wave ordered selection of inside points ----
  {
    const int seg = (N + NWAVES - 1) / NWAVES;
    const int beg = wave * seg;
    const int end = min(beg + seg, N);
    unsigned wc = 0;
    for (int i = beg + lane; i < beg + seg; i += 32) {
      bool inside = false;
      if (i < end) {
        float qx = points[3*i+0], qy = points[3*i+1], qz = points[3*i+2];
        inside = (qx >= x0) & (qx <= x1) & (qy >= y0) & (qy <= y1)
               & (qz >= 0.0f) & (qz <= ah);
      }
      unsigned mask = __builtin_amdgcn_ballot_w32(inside);
      unsigned pre  = __popc(mask & ((1u << lane) - 1u));
      if (inside) {
        unsigned pos = wc + pre;
        if (pos < MAX_PTS) s.u.selW[wave][pos] = i;
      }
      wc += (unsigned)__popc(mask);
    }
    if (lane == 0) s.wcnt[wave] = wc;
  }
  __syncthreads();

  if (tid == 0) {
    unsigned base = 0;
    for (int w = 0; w < NWAVES; ++w) { s.wbase[w] = base; base += s.wcnt[w]; }
    s.total = base;
  }
  __syncthreads();

  const unsigned total = s.total;

  // ---- Gather with wrap semantics: slot i -> selected[i % max(min(total,512),1)] ----
  {
    unsigned m = total < MAX_PTS ? total : MAX_PTS;
    if (m == 0) m = 1;
    for (int slot = tid; slot < MAX_PTS; slot += BLOCK) {
      int pi = 0;  // count==0: reference gathers point 0 (output zeroed by valid)
      if (total > 0) {
        unsigned q = (unsigned)slot % m;
        int w = 0;
        #pragma unroll
        for (int k = 1; k < NWAVES; ++k) if (q >= s.wbase[k]) w = k;
        pi = s.u.selW[w][q - s.wbase[w]];
      }
      s.px[slot] = points[3*pi+0] - ax;
      s.py[slot] = points[3*pi+1] - ay;
      s.pz[slot] = points[3*pi+2];
    }
  }

  // ---- Phase 2+3: two M-halves of 256 rows: F1 (scalar) then F2 via WMMA ----
  float colmax[8];
  #pragma unroll
  for (int t = 0; t < 8; ++t) colmax[t] = 0.0f;

  for (int half = 0; half < 2; ++half) {
    __syncthreads();   // prev consumers of u.f1 / selW done
    const int r0 = half * 256;
    for (int idx = tid; idx < 256*64; idx += BLOCK) {
      int rl = idx >> 6, c = idx & 63;
      int r = r0 + rl;
      float v = s.px[r]*W1[c] + s.py[r]*W1[64+c] + s.pz[r]*W1[128+c] + b1[c];
      v = fmaxf(v, 0.0f);
      s.u.f1[rl*F1_STRIDE + c] = (_Float16)v;
    }
    __syncthreads();

    const int lrow = lane & 15;
    const int kb   = (lane >> 4) << 3;   // 0 or 8 per ISA 16-bit layout
    for (int mt = 0; mt < 2; ++mt) {
      const int m0 = (wave*2 + mt) * 16;
      const _Float16* arow = &s.u.f1[(m0 + lrow)*F1_STRIDE + kb];
      v16h a_lo = ld_frag(arow);         // K = 0..31
      v16h a_hi = ld_frag(arow + 32);    // K = 32..63
      #pragma unroll
      for (int nt = 0; nt < 8; ++nt) {
        const _Float16* bcol = &s.w2t[(nt*16 + lrow)*W2T_STRIDE + kb];
        v16h b_lo = ld_frag(bcol);
        v16h b_hi = ld_frag(bcol + 32);
        v8f acc = {};
        acc = __builtin_amdgcn_wmma_f32_16x16x32_f16(
            false, a_lo, false, b_lo, (short)0, acc, false, false);
        acc = __builtin_amdgcn_wmma_f32_16x16x32_f16(
            false, a_hi, false, b_hi, (short)0, acc, false, false);
        float bias = b2[nt*16 + lrow];
        #pragma unroll
        for (int r = 0; r < 8; ++r) {
          float v = acc[r] + bias;
          v = fmaxf(v, 0.0f);
          colmax[nt] = fmaxf(colmax[nt], v);
        }
      }
    }
  }

  // fold per-lane column maxima into shared (float bits monotonic for >= 0)
  #pragma unroll
  for (int nt = 0; nt < 8; ++nt) {
    int n = nt*16 + (lane & 15);
    atomicMax(&s.gmax[n], __float_as_uint(colmax[nt]));
  }
  __syncthreads();

  // ---- Heads: cls = g@Wc + bc, reg = g@Wr + br ----
  if (tid < 32) {
    float a0 = 0.f, a1 = 0.f, a2 = 0.f, a3 = 0.f, a4 = 0.f;
    for (int k = tid; k < 128; k += 32) {
      float g = __uint_as_float(s.gmax[k]);
      a0 += g * Wc[k];
      a1 += g * Wr[k*4+0];
      a2 += g * Wr[k*4+1];
      a3 += g * Wr[k*4+2];
      a4 += g * Wr[k*4+3];
    }
    s.hpart[tid*5+0] = a0; s.hpart[tid*5+1] = a1; s.hpart[tid*5+2] = a2;
    s.hpart[tid*5+3] = a3; s.hpart[tid*5+4] = a4;
  }
  __syncthreads();
  if (tid == 0) {
    float r[5] = { bc[0], br[0], br[1], br[2], br[3] };
    for (int l = 0; l < 32; ++l)
      #pragma unroll
      for (int j = 0; j < 5; ++j) r[j] += s.hpart[l*5+j];
    float valid = (total >= MIN_PTS) ? 1.0f : 0.0f;
    #pragma unroll
    for (int j = 0; j < 5; ++j) out[anc*5+j] = r[j] * valid;
  }
}

extern "C" void kernel_launch(void* const* d_in, const int* in_sizes, int n_in,
                              void* d_out, int out_size, void* d_ws, size_t ws_size,
                              hipStream_t stream) {
  const float* points  = (const float*)d_in[0];
  const float* anchors = (const float*)d_in[1];
  const float* W1 = (const float*)d_in[2];
  const float* b1 = (const float*)d_in[3];
  const float* W2 = (const float*)d_in[4];
  const float* b2 = (const float*)d_in[5];
  const float* Wc = (const float*)d_in[6];
  const float* bc = (const float*)d_in[7];
  const float* Wr = (const float*)d_in[8];
  const float* br = (const float*)d_in[9];
  float* out = (float*)d_out;

  const int N = in_sizes[0] / 3;
  const int A = in_sizes[1] / 6;

  treercnn_kernel<<<dim3(A), dim3(BLOCK), 0, stream>>>(
      points, anchors, W1, b1, W2, b2, Wc, bc, Wr, br, out, N);
}
